// GKT_25245817766518
// MI455X (gfx1250) — matrix operand order repaired and checked by reading, not compiled, not run on hardware
//
#include <hip/hip_runtime.h>
#include <hip/hip_bf16.h>

typedef __attribute__((ext_vector_type(16))) _Float16 v16h;
typedef __attribute__((ext_vector_type(8)))  float    v8f;

#define B_   64
#define S_   50
#define C_   256
#define HD_  32
#define ED_  32
#define D_   64
#define NROW (B_*C_)   // 16384 rows per step for neigh MLPs
#define NBLK 128       // blocks for row-tiled kernels (128 blocks * 8 waves * 16 rows = 16384)
#define WVB  8         // waves per 256-thread block

__device__ __forceinline__ float sigm(float x) { return 1.0f / (1.0f + expf(-x)); }

__device__ __forceinline__ v8f wmma32(v16h a, v16h b, v8f c) {
    return __builtin_amdgcn_wmma_f32_16x16x32_f16(false, a, false, b, (short)0, c, false, false);
}

__device__ __forceinline__ v16h load_img(const _Float16* imgs, int idx, int lane) {
    return *(const v16h*)(imgs + idx * 512 + lane * 16);
}

// ---------------------------------------------------------------------------
// Weight image packer: builds f16 B-operand register images (one per 32x16
// K/N tile).  B layout (ISA 7.12.2): lane n (0..15) holds column n, halves
// h=0..7 = K 0..7, h=8..15 = K 16..23; lane n+16 holds K 8..15, 24..31 of the same column.
// Images: 0..15  layer1 combined [n0_w1;n1_w1] (K=128 -> 4 ktiles x 4 ntiles)
//         16..17 n0_w2   18..19 n1_w2   20..23 [ea_ew;ea_aw]
//         24..29 gru_wih 30..35 gru_whh
// ---------------------------------------------------------------------------
__global__ void k_prep(_Float16* imgs,
                       const float* n0_w1, const float* n1_w1,
                       const float* n0_w2, const float* n1_w2,
                       const float* ea_ew, const float* ea_aw,
                       const float* wih,   const float* whh)
{
    int img  = blockIdx.x;
    int lane = threadIdx.x;          // 0..31
    int nloc = lane & 15;
    int kb   = (lane < 16) ? 0 : 8;
    _Float16* dst = imgs + img * 512 + lane * 16;
    #pragma unroll
    for (int h = 0; h < 16; ++h) {
        int kk = kb + h + ((h >= 8) ? 8 : 0);
        float val;
        if (img < 16) {                       // layer1 combined, K=128
            int i = img >> 2, j = img & 3;
            int n = j * 16 + nloc;
            int k = i * 32 + kk;
            val = (n < 32) ? n0_w1[n * 128 + k] : n1_w1[(n - 32) * 128 + k];
        } else if (img < 18) {
            int n = (img - 16) * 16 + nloc;
            val = n0_w2[n * 32 + kk];
        } else if (img < 20) {
            int n = (img - 18) * 16 + nloc;
            val = n1_w2[n * 32 + kk];
        } else if (img < 24) {                // [ea_ew ; ea_aw]
            int n = (img - 20) * 16 + nloc;
            val = (n < 32) ? ea_ew[n * 32 + kk] : ea_aw[(n - 32) * 32 + kk];
        } else if (img < 30) {
            int n = (img - 24) * 16 + nloc;
            val = wih[n * 32 + kk];
        } else {
            int n = (img - 30) * 16 + nloc;
            val = whh[n * 32 + kk];
        }
        dst[h] = (_Float16)val;
    }
}

__global__ void k_zero(float* p, int n) {
    int i = blockIdx.x * blockDim.x + threadIdx.x;
    if (i < n) p[i] = 0.0f;
}

// ---------------------------------------------------------------------------
// Kernel 1: per-step self feature.  self_ht[b] = [ht[b,qt], emb_x[xt]] (64),
// fs MLP (64->32 relu, 32->32 relu), batchnorm over the 64 rows.
// One 256-thread block; work is tiny (64 rows).
// ---------------------------------------------------------------------------
__global__ void k_self(int t, const int* features, const int* questions,
                       const float* ht, const float* emb_x,
                       const float* fs_w1, const float* fs_b1,
                       const float* fs_w2, const float* fs_b2,
                       const float* fs_g,  const float* fs_be,
                       float* selfht, float* selffeat)
{
    __shared__ float h1[B_][HD_];
    __shared__ float h2[B_][HD_];
    __shared__ float mu[HD_], rs[HD_];
    int tid = threadIdx.x;
    if (tid < B_) {
        int b  = tid;
        int qt = questions[b * S_ + t];
        int xt = features[b * S_ + t];
        const float* hrow = ht + (b * C_ + qt) * HD_;
        const float* xrow = emb_x + xt * ED_;
        float* dst = selfht + b * D_;
        for (int k = 0; k < HD_; ++k) dst[k]       = hrow[k];
        for (int k = 0; k < ED_; ++k) dst[HD_ + k] = xrow[k];
    }
    __syncthreads();
    for (int e = tid; e < B_ * HD_; e += blockDim.x) {
        int row = e >> 5, col = e & 31;
        const float* x = selfht + row * D_;
        const float* w = fs_w1 + col * D_;
        float acc = fs_b1[col];
        for (int k = 0; k < D_; ++k) acc += x[k] * w[k];
        h1[row][col] = fmaxf(acc, 0.0f);
    }
    __syncthreads();
    for (int e = tid; e < B_ * HD_; e += blockDim.x) {
        int row = e >> 5, col = e & 31;
        const float* w = fs_w2 + col * HD_;
        float acc = fs_b2[col];
        for (int k = 0; k < HD_; ++k) acc += h1[row][k] * w[k];
        h2[row][col] = fmaxf(acc, 0.0f);
    }
    __syncthreads();
    if (tid < HD_) {
        float s = 0.0f, q = 0.0f;
        for (int r = 0; r < B_; ++r) { float v = h2[r][tid]; s += v; q += v * v; }
        float m  = s / (float)B_;
        float vr = q / (float)B_ - m * m;
        mu[tid] = m; rs[tid] = rsqrtf(vr + 1e-5f);
    }
    __syncthreads();
    for (int e = tid; e < B_ * HD_; e += blockDim.x) {
        int row = e >> 5, col = e & 31;
        selffeat[row * HD_ + col] = (h2[row][col] - mu[col]) * rs[col] * fs_g[col] + fs_be[col];
    }
}

// ---------------------------------------------------------------------------
// Kernel 2: neigh MLPs (both n0 & n1 fused) over all 16384 rows via WMMA.
// neigh row = [self_ht[b](64) | ht[b,c](32) | cemb[b,c](32)], K=128.
// Layer1: 16 WMMAs (4 K-tiles x 4 N-tiles, N=64 fused). Layer2: 4 WMMAs.
// Writes pre-norm layer2 outputs (h0s/h1s) + per-block batchnorm partials.
// ---------------------------------------------------------------------------
__global__ void k_neigh(int t, const int* features, const int* questions,
                        const float* ht, const float* emb_x, const float* emb_c,
                        const float* selfht, const _Float16* imgs,
                        const float* n0_b1, const float* n1_b1,
                        const float* n0_b2, const float* n1_b2,
                        float* h0s, float* h1s, float* partials)
{
    __shared__ float lh[WVB][16][64];
    __shared__ float sstats[128];
    int tid  = threadIdx.x;
    int w    = tid >> 5;
    int lane = tid & 31;
    if (tid < 128) sstats[tid] = 0.0f;
    __syncthreads();

    int rt = blockIdx.x * WVB + w;
    int r0 = rt * 16;
    int b  = r0 >> 8;
    int c0 = r0 & 255;
    int m  = lane & 15;
    int c  = c0 + m;
    int koff = (lane < 16) ? 0 : 8;   // A-operand K offset (ISA 7.12.2)
    int ln = lane & 15;
    int mb = (lane < 16) ? 0 : 8;     // C/D row offset
    int qt = questions[b * S_ + t];
    int xt = features[b * S_ + t];

    const float* selfrow = selfht + b * D_;
    const float* htrow   = ht + (b * C_ + c) * HD_;
    const float* cembrow = (c == qt) ? (emb_x + xt * ED_) : (emb_c + c * ED_);

    // ---- layer 1: X(16x128) @ W1c^T(128x64) ----
    v8f acc0 = {}, acc1 = {}, acc2 = {}, acc3 = {};
    #pragma unroll
    for (int i = 0; i < 4; ++i) {
        v16h a;
        #pragma unroll
        for (int half = 0; half < 2; ++half) {
            int ks = i * 32 + koff + half * 16;     // 8-aligned, single segment
            const float* p;
            if (ks < 64)      p = selfrow + ks;
            else if (ks < 96) p = htrow   + (ks - 64);
            else              p = cembrow + (ks - 96);
            #pragma unroll
            for (int j = 0; j < 8; ++j) a[half * 8 + j] = (_Float16)p[j];
        }
        acc0 = wmma32(a, load_img(imgs, i * 4 + 0, lane), acc0);
        acc1 = wmma32(a, load_img(imgs, i * 4 + 1, lane), acc1);
        acc2 = wmma32(a, load_img(imgs, i * 4 + 2, lane), acc2);
        acc3 = wmma32(a, load_img(imgs, i * 4 + 3, lane), acc3);
    }
    #pragma unroll
    for (int r = 0; r < 8; ++r) {           // bias+relu, transpose via LDS
        lh[w][mb + r][ 0 + ln] = fmaxf(acc0[r] + n0_b1[ 0 + ln], 0.0f);
        lh[w][mb + r][16 + ln] = fmaxf(acc1[r] + n0_b1[16 + ln], 0.0f);
        lh[w][mb + r][32 + ln] = fmaxf(acc2[r] + n1_b1[ 0 + ln], 0.0f);
        lh[w][mb + r][48 + ln] = fmaxf(acc3[r] + n1_b1[16 + ln], 0.0f);
    }
    __syncthreads();

    // ---- layer 2: h0 @ n0_w2^T, h1 @ n1_w2^T ----
    v16h a0, a1;
    {
        const float* row0 = &lh[w][m][0];
        const float* row1 = &lh[w][m][32];
        #pragma unroll
        for (int j = 0; j < 8; ++j) {
            a0[j]     = (_Float16)row0[koff + j];
            a0[8 + j] = (_Float16)row0[koff + 16 + j];
            a1[j]     = (_Float16)row1[koff + j];
            a1[8 + j] = (_Float16)row1[koff + 16 + j];
        }
    }
    v8f d0 = {}, d1 = {}, d2 = {}, d3 = {};
    d0 = wmma32(a0, load_img(imgs, 16, lane), d0);
    d1 = wmma32(a0, load_img(imgs, 17, lane), d1);
    d2 = wmma32(a1, load_img(imgs, 18, lane), d2);
    d3 = wmma32(a1, load_img(imgs, 19, lane), d3);

    float s0 = 0, q0 = 0, s1 = 0, q1 = 0, s2 = 0, q2 = 0, s3 = 0, q3 = 0;
    #pragma unroll
    for (int r = 0; r < 8; ++r) {
        int rr = r0 + mb + r;
        float v0 = fmaxf(d0[r] + n0_b2[ 0 + ln], 0.0f);
        float v1 = fmaxf(d1[r] + n0_b2[16 + ln], 0.0f);
        float v2 = fmaxf(d2[r] + n1_b2[ 0 + ln], 0.0f);
        float v3 = fmaxf(d3[r] + n1_b2[16 + ln], 0.0f);
        h0s[rr * 32 +  0 + ln] = v0;
        h0s[rr * 32 + 16 + ln] = v1;
        h1s[rr * 32 +  0 + ln] = v2;
        h1s[rr * 32 + 16 + ln] = v3;
        s0 += v0; q0 += v0 * v0;  s1 += v1; q1 += v1 * v1;
        s2 += v2; q2 += v2 * v2;  s3 += v3; q3 += v3 * v3;
    }
    atomicAdd(&sstats[ 0 +  0 + ln], s0); atomicAdd(&sstats[32 +  0 + ln], q0);
    atomicAdd(&sstats[ 0 + 16 + ln], s1); atomicAdd(&sstats[32 + 16 + ln], q1);
    atomicAdd(&sstats[64 +  0 + ln], s2); atomicAdd(&sstats[96 +  0 + ln], q2);
    atomicAdd(&sstats[64 + 16 + ln], s3); atomicAdd(&sstats[96 + 16 + ln], q3);
    __syncthreads();
    if (tid < 128) partials[blockIdx.x * 128 + tid] = sstats[tid];
}

// ---------------------------------------------------------------------------
// Kernel 3: finalize batchnorm stats (mean / rsqrt(var+eps)) for n0 and n1.
// stats: [0..31]=mu0 [32..63]=rstd0 [64..95]=mu1 [96..127]=rstd1
// ---------------------------------------------------------------------------
__global__ void k_stats(const float* partials, float* stats)
{
    int tid = threadIdx.x;               // 64 threads
    int mlp = tid >> 5, col = tid & 31;
    float s = 0.0f, q = 0.0f;
    for (int blk = 0; blk < NBLK; ++blk) {
        s += partials[blk * 128 + mlp * 64 + col];
        q += partials[blk * 128 + mlp * 64 + 32 + col];
    }
    float invN = 1.0f / (float)NROW;
    float mu   = s * invN;
    float var  = q * invN - mu * mu;
    stats[mlp * 64 + col]      = mu;
    stats[mlp * 64 + 32 + col] = rsqrtf(var + 1e-5f);
}

// ---------------------------------------------------------------------------
// Kernel 4: norm + adjacency combine + erase/add (WMMA 32->64) + GRU
// (WMMA 32->96 x2) + in-place ht update + prediction gather.
// ---------------------------------------------------------------------------
__global__ void k_update(int t, const int* questions,
                         float* ht, const float* selffeat,
                         const float* h0s, const float* h1s, const float* stats,
                         const float* n0_g, const float* n0_be,
                         const float* n1_g, const float* n1_be,
                         const float* graph, const float* ea_w,
                         const float* ea_eb, const float* ea_ab,
                         const _Float16* imgs,
                         const float* gru_bih, const float* gru_bhh,
                         const float* pr_w, const float* pr_b, float* out)
{
    __shared__ float lht[WVB][16][32];
    __shared__ float lm[WVB][16][32];
    int tid  = threadIdx.x;
    int w    = tid >> 5;
    int lane = tid & 31;
    int rt = blockIdx.x * WVB + w;
    int r0 = rt * 16;
    int b  = r0 >> 8;
    int c0 = r0 & 255;
    int ln = lane & 15;
    int koff = (lane < 16) ? 0 : 8;
    int mb   = (lane < 16) ? 0 : 8;
    int qt  = questions[b * S_ + t];
    int qtn = questions[b * S_ + t + 1];

    // load ht tile; build m_next tile (norm + adjacency + self override)
    #pragma unroll
    for (int i = 0; i < 16; ++i) {
        int e = lane * 16 + i;
        int mm = e >> 5, n = e & 31;
        int c  = c0 + mm;
        lht[w][mm][n] = ht[(b * C_ + c) * HD_ + n];
        float f0 = (h0s[(r0 + mm) * 32 + n] - stats[n])      * stats[32 + n] * n0_g[n] + n0_be[n];
        float f1 = (h1s[(r0 + mm) * 32 + n] - stats[64 + n]) * stats[96 + n] * n1_g[n] + n1_be[n];
        float adj  = graph[qt * C_ + c];
        float radj = graph[c * C_ + qt];
        float mv = adj * f0 + radj * f1;
        if (c == qt) mv = selffeat[b * HD_ + n];
        lm[w][mm][n] = mv;
    }
    __syncthreads();

    // ---- erase/add: [er|ad] = m @ [ea_ew;ea_aw]^T ----
    v16h am;
    {
        const float* row = &lm[w][ln][0];
        #pragma unroll
        for (int j = 0; j < 8; ++j) {
            am[j]     = (_Float16)row[koff + j];
            am[8 + j] = (_Float16)row[koff + 16 + j];
        }
    }
    v8f e0 = {}, e1 = {}, a0 = {}, a1 = {};
    e0 = wmma32(am, load_img(imgs, 20, lane), e0);
    e1 = wmma32(am, load_img(imgs, 21, lane), e1);
    a0 = wmma32(am, load_img(imgs, 22, lane), a0);
    a1 = wmma32(am, load_img(imgs, 23, lane), a1);
    #pragma unroll
    for (int r = 0; r < 8; ++r) {
        int mr = mb + r;
        float wc = ea_w[c0 + mr];
        {
            int n = ln;
            float mv = lm[w][mr][n];
            float er = sigm(e0[r] + ea_eb[n]);
            float av = tanhf(a0[r] + ea_ab[n]);
            lm[w][mr][n] = mv - wc * er * mv + wc * av;
        }
        {
            int n = 16 + ln;
            float mv = lm[w][mr][n];
            float er = sigm(e1[r] + ea_eb[n]);
            float av = tanhf(a1[r] + ea_ab[n]);
            lm[w][mr][n] = mv - wc * er * mv + wc * av;
        }
    }
    __syncthreads();

    // ---- GRU: gi = m2 @ wih^T (32->96), gh = ht @ whh^T (32->96) ----
    v16h am2, ah;
    {
        const float* rm = &lm[w][ln][0];
        const float* rh = &lht[w][ln][0];
        #pragma unroll
        for (int j = 0; j < 8; ++j) {
            am2[j]     = (_Float16)rm[koff + j];
            am2[8 + j] = (_Float16)rm[koff + 16 + j];
            ah[j]      = (_Float16)rh[koff + j];
            ah[8 + j]  = (_Float16)rh[koff + 16 + j];
        }
    }
    v8f gi0 = {}, gi1 = {}, gi2 = {}, gi3 = {}, gi4 = {}, gi5 = {};
    v8f gh0 = {}, gh1 = {}, gh2 = {}, gh3 = {}, gh4 = {}, gh5 = {};
    gi0 = wmma32(am2, load_img(imgs, 24, lane), gi0);
    gi1 = wmma32(am2, load_img(imgs, 25, lane), gi1);
    gi2 = wmma32(am2, load_img(imgs, 26, lane), gi2);
    gi3 = wmma32(am2, load_img(imgs, 27, lane), gi3);
    gi4 = wmma32(am2, load_img(imgs, 28, lane), gi4);
    gi5 = wmma32(am2, load_img(imgs, 29, lane), gi5);
    gh0 = wmma32(ah,  load_img(imgs, 30, lane), gh0);
    gh1 = wmma32(ah,  load_img(imgs, 31, lane), gh1);
    gh2 = wmma32(ah,  load_img(imgs, 32, lane), gh2);
    gh3 = wmma32(ah,  load_img(imgs, 33, lane), gh3);
    gh4 = wmma32(ah,  load_img(imgs, 34, lane), gh4);
    gh5 = wmma32(ah,  load_img(imgs, 35, lane), gh5);
    __syncthreads();

    #pragma unroll
    for (int r = 0; r < 8; ++r) {
        int mr = mb + r;
        int c  = c0 + mr;
        #pragma unroll
        for (int j2 = 0; j2 < 2; ++j2) {
            int n = j2 * 16 + ln;
            float ir, iz, in_, hr, hz, hn;
            if (j2 == 0) { ir = gi0[r]; iz = gi2[r]; in_ = gi4[r]; hr = gh0[r]; hz = gh2[r]; hn = gh4[r]; }
            else         { ir = gi1[r]; iz = gi3[r]; in_ = gi5[r]; hr = gh1[r]; hz = gh3[r]; hn = gh5[r]; }
            ir  += gru_bih[n];       hr += gru_bhh[n];
            iz  += gru_bih[32 + n];  hz += gru_bhh[32 + n];
            in_ += gru_bih[64 + n];  hn += gru_bhh[64 + n];
            float rg = sigm(ir + hr);
            float z  = sigm(iz + hz);
            float nn = tanhf(in_ + rg * hn);
            float hp = lht[w][mr][n];
            float hnew = (1.0f - z) * nn + z * hp;
            ht[(b * C_ + c) * HD_ + n] = hnew;
            lht[w][mr][n] = hnew;
        }
    }
    __syncthreads();

    if (lane < 16) {
        int c = c0 + lane;
        if (c == qtn) {
            float acc = pr_b[0];
            for (int k = 0; k < HD_; ++k) acc += lht[w][lane][k] * pr_w[k];
            out[b * (S_ - 1) + t] = sigm(acc);
        }
    }
}

// ---------------------------------------------------------------------------
extern "C" void kernel_launch(void* const* d_in, const int* in_sizes, int n_in,
                              void* d_out, int out_size, void* d_ws, size_t ws_size,
                              hipStream_t stream)
{
    (void)in_sizes; (void)n_in; (void)out_size; (void)ws_size;
    const int*   features  = (const int*)d_in[0];
    const int*   questions = (const int*)d_in[1];
    const float* graph     = (const float*)d_in[2];
    const float* emb_x     = (const float*)d_in[3];
    const float* emb_c     = (const float*)d_in[4];
    const float* fs_w1 = (const float*)d_in[5];
    const float* fs_b1 = (const float*)d_in[6];
    const float* fs_w2 = (const float*)d_in[7];
    const float* fs_b2 = (const float*)d_in[8];
    const float* fs_g  = (const float*)d_in[9];
    const float* fs_be = (const float*)d_in[10];
    const float* n0_w1 = (const float*)d_in[11];
    const float* n0_b1 = (const float*)d_in[12];
    const float* n0_w2 = (const float*)d_in[13];
    const float* n0_b2 = (const float*)d_in[14];
    const float* n0_g  = (const float*)d_in[15];
    const float* n0_be = (const float*)d_in[16];
    const float* n1_w1 = (const float*)d_in[17];
    const float* n1_b1 = (const float*)d_in[18];
    const float* n1_w2 = (const float*)d_in[19];
    const float* n1_b2 = (const float*)d_in[20];
    const float* n1_g  = (const float*)d_in[21];
    const float* n1_be = (const float*)d_in[22];
    const float* ea_w  = (const float*)d_in[23];
    const float* ea_ew = (const float*)d_in[24];
    const float* ea_eb = (const float*)d_in[25];
    const float* ea_aw = (const float*)d_in[26];
    const float* ea_ab = (const float*)d_in[27];
    const float* gru_wih = (const float*)d_in[28];
    const float* gru_whh = (const float*)d_in[29];
    const float* gru_bih = (const float*)d_in[30];
    const float* gru_bhh = (const float*)d_in[31];
    const float* pr_w  = (const float*)d_in[32];
    const float* pr_b  = (const float*)d_in[33];
    float* out = (float*)d_out;

    float* ws       = (float*)d_ws;
    float* ht       = ws;                       // 524288 floats
    float* selfht   = ht + B_ * C_ * HD_;       // 4096
    float* selffeat = selfht + B_ * D_;         // 2048
    float* h0s      = selffeat + B_ * HD_;      // 524288
    float* h1s      = h0s + NROW * HD_;         // 524288
    float* partials = h1s + NROW * HD_;         // 16384
    float* stats    = partials + NBLK * 128;    // 128
    _Float16* imgs  = reinterpret_cast<_Float16*>(stats + 128); // 36*512 halves

    k_prep<<<36, 32, 0, stream>>>(imgs, n0_w1, n1_w1, n0_w2, n1_w2, ea_ew, ea_aw, gru_wih, gru_whh);
    k_zero<<<(B_ * C_ * HD_ + 255) / 256, 256, 0, stream>>>(ht, B_ * C_ * HD_);

    for (int t = 0; t < S_ - 1; ++t) {
        k_self<<<1, 256, 0, stream>>>(t, features, questions, ht, emb_x,
                                      fs_w1, fs_b1, fs_w2, fs_b2, fs_g, fs_be,
                                      selfht, selffeat);
        k_neigh<<<NBLK, 256, 0, stream>>>(t, features, questions, ht, emb_x, emb_c,
                                          selfht, imgs, n0_b1, n1_b1, n0_b2, n1_b2,
                                          h0s, h1s, partials);
        k_stats<<<1, 64, 0, stream>>>(partials, stats);
        k_update<<<NBLK, 256, 0, stream>>>(t, questions, ht, selffeat, h0s, h1s, stats,
                                           n0_g, n0_be, n1_g, n1_be, graph,
                                           ea_w, ea_eb, ea_ab, imgs,
                                           gru_bih, gru_bhh, pr_w, pr_b, out);
    }
}
